// GAttn_59889023975886
// MI455X (gfx1250) — matrix operand order repaired and checked by reading, not compiled, not run on hardware
//
#include <hip/hip_runtime.h>

// ---------------------------------------------------------------------------
// Types for CDNA5 WMMA (gfx1250, wave32)
// ---------------------------------------------------------------------------
typedef __attribute__((ext_vector_type(16))) __bf16 v16bf;
typedef __attribute__((ext_vector_type(8)))  __bf16 v8bf;
typedef __attribute__((ext_vector_type(8)))  float  v8f;

__device__ __forceinline__ v8f wmma_bf16(v16bf a, v16bf b, v8f c) {
  // D(16x16 f32) = A(16x32 bf16) * B(32x16 bf16) + C
  return __builtin_amdgcn_wmma_f32_16x16x32_bf16(false, a, false, b,
                                                 (short)0, c, false, false);
}

__device__ __forceinline__ __bf16 f2bf(float f) {
  unsigned u = __builtin_bit_cast(unsigned, f);
  unsigned r = (u + 0x7FFFu + ((u >> 16) & 1u)) >> 16;
  return __builtin_bit_cast(__bf16, (unsigned short)r);
}
__device__ __forceinline__ float bf2f(__bf16 h) {
  unsigned u = ((unsigned)__builtin_bit_cast(unsigned short, h)) << 16;
  return __builtin_bit_cast(float, u);
}
__device__ __forceinline__ float sigmoidf_(float z) {
  return 1.0f / (1.0f + __expf(-z));
}

#define NPIX   6400      // 80*80
#define CCH    128
#define HEADS  4
#define HD     32
#define NA     1600      // NPIX / AREA
#define BE     8         // B * AREA
#define MTOT   12800     // B * NPIX

// ---------------------------------------------------------------------------
// Kernel 1: transpose weights to (K x OC) bf16.  wcatT: 128 x 384
//   cols [0,256) = w_qk rows, cols [256,384) = w_in rows.   woutT: 128 x 128
// ---------------------------------------------------------------------------
__global__ __launch_bounds__(256) void prep_weights(
    const float* __restrict__ w_qk, const float* __restrict__ w_in,
    const float* __restrict__ w_out, __bf16* __restrict__ wcatT,
    __bf16* __restrict__ woutT) {
  int idx = blockIdx.x * blockDim.x + threadIdx.x;
  if (idx < 128 * 384) {
    int k = idx / 384, oc = idx % 384;
    float v = (oc < 256) ? w_qk[oc * 128 + k] : w_in[(oc - 256) * 128 + k];
    wcatT[idx] = f2bf(v);
  } else {
    int idx2 = idx - 128 * 384;
    if (idx2 < 128 * 128) {
      int k = idx2 / 128, oc = idx2 % 128;
      woutT[idx2] = f2bf(w_out[oc * 128 + k]);
    }
  }
}

// ---------------------------------------------------------------------------
// Kernel 2: NCHW fp32 -> pixel-major bf16 (LDS-tiled transpose)
// ---------------------------------------------------------------------------
__global__ __launch_bounds__(256) void transpose_x(const float* __restrict__ x,
                                                   __bf16* __restrict__ xT) {
  __shared__ float tile[32][33];
  const int b = blockIdx.z, c0 = blockIdx.y * 32, n0 = blockIdx.x * 32;
  const int tx = threadIdx.x, ty = threadIdx.y;  // 32 x 8
#pragma unroll
  for (int k = 0; k < 4; ++k) {
    int c = c0 + ty + k * 8;
    tile[ty + k * 8][tx] = x[((size_t)b * CCH + c) * NPIX + n0 + tx];
  }
  __syncthreads();
#pragma unroll
  for (int k = 0; k < 4; ++k) {
    int n = n0 + ty + k * 8;
    xT[((size_t)b * NPIX + n) * CCH + c0 + tx] = f2bf(tile[tx][ty + k * 8]);
  }
}

// ---------------------------------------------------------------------------
// Kernel 3: QKV projection GEMM.  A = xT (12800 x 128), B = wcatT (128 x 384).
// One wave per 16x16 output tile; 4 WMMA K-steps.  Scalar (wave-uniform)
// tile routing: oc0<128 -> Q, oc0<256 -> K (transposed), else V.
// ---------------------------------------------------------------------------
__global__ __launch_bounds__(256) void gemm_qkv(
    const __bf16* __restrict__ xT, const __bf16* __restrict__ wcatT,
    const float* __restrict__ s_qk, const float* __restrict__ t_qk,
    const float* __restrict__ s_in, const float* __restrict__ t_in,
    __bf16* __restrict__ Qo, __bf16* __restrict__ Ko, __bf16* __restrict__ Vo) {
  const int lane = threadIdx.x & 31;
  const int wv = __builtin_amdgcn_readfirstlane(threadIdx.x >> 5);
  const int wid = blockIdx.x * 8 + wv;         // 0..19199  (wave-uniform)
  const int mt = wid / 24, ot = wid % 24;
  const int pix0 = mt * 16, oc0 = ot * 16;
  const int hgrp = lane >> 4, l16 = lane & 15;

  v8f acc = {};
#pragma unroll
  for (int kk = 0; kk < 4; ++kk) {
    v16bf af;
    {
      const __bf16* ap = xT + (size_t)(pix0 + l16) * CCH + kk * 32 + (hgrp ? 8 : 0);
      v8bf lo = *(const v8bf*)(ap);
      v8bf hi = *(const v8bf*)(ap + 16);
#pragma unroll
      for (int e = 0; e < 8; ++e) { af[e] = lo[e]; af[8 + e] = hi[e]; }
    }
    v16bf bfm;
#pragma unroll
    for (int s = 0; s < 16; ++s)
      bfm[s] = wcatT[(size_t)(kk * 32 + ((s & 8) << 1) + hgrp * 8 + (s & 7)) * 384 + oc0 + l16];
    acc = wmma_bf16(af, bfm, acc);
  }

  // tile-uniform pixel decomposition (1600 | 16, so be/i split is uniform)
  const int be0 = pix0 / NA;       // scalar div by constant
  const int ir0 = pix0 % NA;
  const float RSQRT_HD = 0.17677669529663687f;  // 1/sqrt(32)

  if (oc0 < 128) {                 // ---- Q tile ----
    int hh = oc0 >> 5;
    int dd = (oc0 & 31) + l16;
    float sc = s_qk[oc0 + l16], tc = t_qk[oc0 + l16];
    __bf16* qbase = Qo + ((size_t)(be0 * HEADS + hh) * NA + ir0) * HD + dd;
#pragma unroll
    for (int r = 0; r < 8; ++r) {
      int m = hgrp ? (8 + r) : r;
      float d = acc[r] * sc + tc;
      qbase[(size_t)m * HD] = f2bf(d * RSQRT_HD);
    }
  } else if (oc0 < 256) {          // ---- K tile (transposed) ----
    int kc0 = oc0 - 128;
    int hh = kc0 >> 5;
    int dd = (kc0 & 31) + l16;
    float sc = s_qk[oc0 + l16], tc = t_qk[oc0 + l16];
    __bf16* kbase = Ko + ((size_t)(be0 * HEADS + hh) * HD + dd) * NA + ir0;
#pragma unroll
    for (int r = 0; r < 8; ++r) {
      int m = hgrp ? (8 + r) : r;
      float d = acc[r] * sc + tc;
      kbase[m] = f2bf(d);
    }
  } else {                         // ---- V tile ----
    int vc0 = oc0 - 256;
    int hh = vc0 >> 5;
    int dd = (vc0 & 31) + l16;
    float sc = s_in[vc0 + l16], tc = t_in[vc0 + l16];
    __bf16* vbase = Vo + ((size_t)(be0 * HEADS + hh) * NA + ir0) * HD + dd;
    const __bf16* xbase = xT + (size_t)pix0 * CCH + vc0 + l16;
#pragma unroll
    for (int r = 0; r < 8; ++r) {
      int m = hgrp ? (8 + r) : r;
      float d = acc[r] * sc + tc;
      float g = sigmoidf_(d);
      float xv = bf2f(xbase[(size_t)m * CCH]);
      vbase[(size_t)m * HD] = f2bf(g * xv);
    }
  }
}

// ---------------------------------------------------------------------------
// Kernel 4: fused area attention.
// Block = (be, 16-query tile), 4 waves = 4 heads.
// Pass 1: score WMMAs + lane-local online softmax stats (no shuffles in the
//   hot loop), single cross-lane (m,l) merge at the end.
// Pass 2: recompute score tiles, normalize, 4-head 1x5 conv through a
//   power-of-two 128-column LDS ring (index = j & 127, halo columns zeroed),
//   SiLU, feed straight back into WMMA against V.
// The 8x4x1600x1600 attention matrix never touches memory.
// ---------------------------------------------------------------------------
__global__ __launch_bounds__(128) void attn_kernel(
    const __bf16* __restrict__ Q, const __bf16* __restrict__ K,
    const __bf16* __restrict__ V, __bf16* __restrict__ O,
    const float* __restrict__ w_fg, const float* __restrict__ s_fg,
    const float* __restrict__ t_fg) {
  __shared__ float p_lds[HEADS * 16 * 128];  // [head][i][j & 127], 32 KB

  const int lane = threadIdx.x & 31;
  const int sh   = __builtin_amdgcn_readfirstlane(threadIdx.x >> 5);  // head
  const int be   = __builtin_amdgcn_readfirstlane(blockIdx.x) / (NA / 16);
  const int i0   = (__builtin_amdgcn_readfirstlane(blockIdx.x) % (NA / 16)) * 16;
  const int hgrp = lane >> 4, l16 = lane & 15;

  // zero the ring so halo reads (j<0 wrap) see exact zeros
  for (int i = threadIdx.x; i < HEADS * 16 * 128; i += 128) p_lds[i] = 0.f;
  __syncthreads();

  // conv weights for this output head
  float wreg[4][5];
#pragma unroll
  for (int hin = 0; hin < 4; ++hin)
#pragma unroll
    for (int dk = 0; dk < 5; ++dk)
      wreg[hin][dk] = w_fg[(sh * 4 + hin) * 5 + dk];
  const float sfg = s_fg[sh], tfg = t_fg[sh];

  const __bf16* Qh = Q + (size_t)(be * HEADS + sh) * NA * HD;
  const __bf16* Kh = K + (size_t)(be * HEADS + sh) * HD * NA;
  const __bf16* Vh = V + (size_t)(be * HEADS + sh) * NA * HD;

  // Q A-fragment (16 x 32), reused for both passes
  v16bf qf;
  {
    const __bf16* qp = Qh + (size_t)(i0 + l16) * HD + (hgrp ? 8 : 0);
    v8bf lo = *(const v8bf*)(qp);
    v8bf hi = *(const v8bf*)(qp + 16);
#pragma unroll
    for (int e = 0; e < 8; ++e) { qf[e] = lo[e]; qf[8 + e] = hi[e]; }
  }

  // ---- Pass 1: lane-local online row stats over all 100 key tiles ----
  float mrow[8], lrow[8];
#pragma unroll
  for (int r = 0; r < 8; ++r) { mrow[r] = -3.0e38f; lrow[r] = 0.f; }

  for (int jt = 0; jt < NA / 16; ++jt) {
    const __bf16* kp = Kh + jt * 16 + l16;
    __builtin_prefetch((const void*)(kp + 16 * 8), 0, 1);  // ~8 tiles ahead
    v16bf kf;
#pragma unroll
    for (int s = 0; s < 16; ++s)
      kf[s] = kp[(size_t)(((s & 8) << 1) + hgrp * 8 + (s & 7)) * NA];
    v8f acc = {};
    acc = wmma_bf16(qf, kf, acc);
#pragma unroll
    for (int r = 0; r < 8; ++r) {
      float v = acc[r];
      float mn = fmaxf(mrow[r], v);
      lrow[r] = lrow[r] * __expf(mrow[r] - mn) + __expf(v - mn);
      mrow[r] = mn;
    }
  }
  // cross-lane merge of (m,l) pairs within each 16-lane row group
  float invl[8];
#pragma unroll
  for (int r = 0; r < 8; ++r) {
    float m = mrow[r], l = lrow[r];
#pragma unroll
    for (int off = 8; off; off >>= 1) {
      float mo = __shfl_xor(m, off, 16);
      float lo = __shfl_xor(l, off, 16);
      float mn = fmaxf(m, mo);
      l = l * __expf(m - mn) + lo * __expf(mo - mn);
      m = mn;
    }
    mrow[r] = m;
    invl[r] = 1.0f / l;
  }

  // ---- Pass 2: stream 32-wide chunks through the 128-column ring ----
  v8f accO0 = {}, accO1 = {};

  for (int t = 0; t <= NA / 32; ++t) {  // 0..50
    if (t < NA / 32) {
      const int colbase = (t * 32) & 127;  // scalar
#pragma unroll
      for (int s = 0; s < 2; ++s) {
        int j0 = t * 32 + s * 16;
        v16bf kf;
#pragma unroll
        for (int q = 0; q < 16; ++q)
          kf[q] = Kh[(size_t)(((q & 8) << 1) + hgrp * 8 + (q & 7)) * NA + j0 + l16];
        v8f acc = {};
        acc = wmma_bf16(qf, kf, acc);
#pragma unroll
        for (int r = 0; r < 8; ++r) {
          float p = __expf(acc[r] - mrow[r]) * invl[r];
          int m = hgrp ? (8 + r) : r;
          p_lds[(sh * 16 + m) * 128 + colbase + s * 16 + l16] = p;
        }
      }
    } else {
      // t == 50: zero right-halo columns (global j = 1600, 1601 -> col 64,65)
      p_lds[(sh * 16 + l16) * 128 + 64 + hgrp] = 0.f;
    }
    __syncthreads();
    if (t >= 1) {
      const int c = t - 1;
      const int m = l16;
      v16bf ff = {};
#pragma unroll
      for (int blk = 0; blk < 2; ++blk) {
        const int kb = (hgrp ? 8 : 0) + blk * 16;  // 8-wide run start (local k)
        float win[4][12];
#pragma unroll
        for (int hin = 0; hin < 4; ++hin) {
          const float* row = p_lds + (hin * 16 + m) * 128;
#pragma unroll
          for (int w = 0; w < 12; ++w) {
            int jd = c * 32 + kb - 2 + w;
            win[hin][w] = row[jd & 127];   // halos are exact zeros
          }
        }
#pragma unroll
        for (int e = 0; e < 8; ++e) {
          float sum = 0.f;
#pragma unroll
          for (int hin = 0; hin < 4; ++hin)
#pragma unroll
            for (int dk = 0; dk < 5; ++dk)
              sum = __builtin_fmaf(wreg[hin][dk], win[hin][e + dk], sum);
          float z = sum * sfg + tfg;
          ff[blk * 8 + e] = f2bf(z * sigmoidf_(z));  // silu
        }
      }
      // O^T tile (16 i x 32 d) += f (16 x 32j) * V^T (32j x 32d)
      const __bf16* vb = Vh + (size_t)(c * 32) * HD + l16;
      v16bf v0, v1;
#pragma unroll
      for (int s = 0; s < 16; ++s) {
        size_t ko = (size_t)(((s & 8) << 1) + hgrp * 8 + (s & 7)) * HD;
        v0[s] = vb[ko];
        v1[s] = vb[ko + 16];
      }
      accO0 = wmma_bf16(ff, v0, accO0);
      accO1 = wmma_bf16(ff, v1, accO1);
    }
    __syncthreads();
  }

  // store O pixel-major bf16: O[pix*128 + h*32 + d]
  const int bIdx = be >> 2, area = be & 3;
  const size_t pixBase = (size_t)bIdx * NPIX + (size_t)area * NA + i0;
#pragma unroll
  for (int r = 0; r < 8; ++r) {
    int m = hgrp ? (8 + r) : r;
    size_t pix = pixBase + m;
    int cidx = sh * HD + l16;
    O[pix * CCH + cidx]      = f2bf(accO0[r]);
    O[pix * CCH + cidx + 16] = f2bf(accO1[r]);
  }
}

// ---------------------------------------------------------------------------
// Kernel 5: output GEMM + sigmoid gate + residual, writes NCHW fp32.
// ---------------------------------------------------------------------------
__global__ __launch_bounds__(256) void gemm_out(
    const __bf16* __restrict__ Ob, const __bf16* __restrict__ woutT,
    const float* __restrict__ s_out, const float* __restrict__ t_out,
    const float* __restrict__ x, float* __restrict__ out) {
  const int lane = threadIdx.x & 31;
  const int wv = __builtin_amdgcn_readfirstlane(threadIdx.x >> 5);
  const int wid = blockIdx.x * 8 + wv;  // 0..6399 (wave-uniform)
  const int mt = wid / 8, ot = wid % 8;
  const int pix0 = mt * 16, oc0 = ot * 16;
  const int hgrp = lane >> 4, l16 = lane & 15;

  v8f acc = {};
#pragma unroll
  for (int kk = 0; kk < 4; ++kk) {
    v16bf af;
    {
      const __bf16* ap = Ob + (size_t)(pix0 + l16) * CCH + kk * 32 + (hgrp ? 8 : 0);
      v8bf lo = *(const v8bf*)(ap);
      v8bf hi = *(const v8bf*)(ap + 16);
#pragma unroll
      for (int e = 0; e < 8; ++e) { af[e] = lo[e]; af[8 + e] = hi[e]; }
    }
    v16bf bfm;
#pragma unroll
    for (int s = 0; s < 16; ++s)
      bfm[s] = woutT[(size_t)(kk * 32 + ((s & 8) << 1) + hgrp * 8 + (s & 7)) * CCH + oc0 + l16];
    acc = wmma_bf16(af, bfm, acc);
  }

  // tile-uniform decomposition (6400 | 16)
  const int b0 = pix0 / NPIX;
  const int n0 = pix0 % NPIX;
  const int oc = oc0 + l16;
  const float sc = s_out[oc], tc = t_out[oc];
  const float* xbase = x + ((size_t)b0 * CCH + oc) * NPIX + n0;
  float* obase = out + ((size_t)b0 * CCH + oc) * NPIX + n0;
  const __bf16* obb = Ob + (size_t)pix0 * CCH + oc;
#pragma unroll
  for (int r = 0; r < 8; ++r) {
    int m = hgrp ? (8 + r) : r;
    float d = acc[r] * sc + tc;
    float g = sigmoidf_(d);
    float ov = bf2f(obb[(size_t)m * CCH]);
    obase[m] = g * ov + xbase[m];
  }
}

// ---------------------------------------------------------------------------
// Host launcher
// ---------------------------------------------------------------------------
extern "C" void kernel_launch(void* const* d_in, const int* in_sizes, int n_in,
                              void* d_out, int out_size, void* d_ws, size_t ws_size,
                              hipStream_t stream) {
  const float* x     = (const float*)d_in[0];
  const float* w_qk  = (const float*)d_in[1];
  const float* s_qk  = (const float*)d_in[2];
  const float* t_qk  = (const float*)d_in[3];
  const float* w_in  = (const float*)d_in[4];
  const float* s_in  = (const float*)d_in[5];
  const float* t_in  = (const float*)d_in[6];
  const float* w_out = (const float*)d_in[7];
  const float* s_out = (const float*)d_in[8];
  const float* t_out = (const float*)d_in[9];
  const float* w_fg  = (const float*)d_in[10];
  const float* s_fg  = (const float*)d_in[11];
  const float* t_fg  = (const float*)d_in[12];
  float* out = (float*)d_out;

  char* ws = (char*)d_ws;
  auto alloc = [&](size_t bytes) {
    char* p = ws;
    ws += (bytes + 255) & ~(size_t)255;
    return p;
  };
  __bf16* xT    = (__bf16*)alloc((size_t)MTOT * CCH * 2);
  __bf16* wcatT = (__bf16*)alloc((size_t)128 * 384 * 2);
  __bf16* woutT = (__bf16*)alloc((size_t)128 * 128 * 2);
  __bf16* Qb    = (__bf16*)alloc((size_t)BE * HEADS * NA * HD * 2);
  __bf16* Kb    = (__bf16*)alloc((size_t)BE * HEADS * HD * NA * 2);
  __bf16* Vb    = (__bf16*)alloc((size_t)BE * HEADS * NA * HD * 2);
  __bf16* Ob    = (__bf16*)alloc((size_t)MTOT * CCH * 2);

  prep_weights<<<dim3(256), dim3(256), 0, stream>>>(w_qk, w_in, w_out, wcatT, woutT);
  transpose_x<<<dim3(NPIX / 32, CCH / 32, 2), dim3(32, 8), 0, stream>>>(x, xT);
  gemm_qkv<<<dim3((MTOT / 16) * 24 / 8), dim3(256), 0, stream>>>(
      xT, wcatT, s_qk, t_qk, s_in, t_in, Qb, Kb, Vb);
  attn_kernel<<<dim3(BE * (NA / 16)), dim3(128), 0, stream>>>(
      Qb, Kb, Vb, Ob, w_fg, s_fg, t_fg);
  gemm_out<<<dim3((MTOT / 16) * 8 / 8), dim3(256), 0, stream>>>(
      Ob, woutT, s_out, t_out, x, out);
}
